// MaskEnc_90615220011190
// MI455X (gfx1250) — matrix operand order repaired and checked by reading, not compiled, not run on hardware
//
#include <hip/hip_runtime.h>
#include <cstdint>

typedef __bf16 bf16_t;
typedef __attribute__((ext_vector_type(16))) __bf16 bf16x16;
typedef __attribute__((ext_vector_type(8)))  float  f32x8;

#define B_SZ 4
#define N_SZ 1024
#define D_SZ 1024
#define H_SZ 16
#define DH_SZ 64
#define Z_SZ 64                 // H*B head-major batch
#define MROWS (B_SZ * N_SZ)     // 4096
#define NEG_BIG (-4294967295.0f)

// epilogue modes
#define EPI_BF16            0
#define EPI_BF16_HEADS      1
#define EPI_SCORES          2
#define EPI_F32_HEADMERGE   3
#define EPI_BF16_BIAS_RELU  4
#define EPI_F32_BIAS        5

// GEMM tiling: 8 waves (wave32), block tile 128x64, 32x32 per wave
#define BM 128
#define BN 64
#define BK 32
#define LSTR 40                 // LDS row stride in bf16 (32 + 8 pad, keeps 16B align)

// ---------------------------------------------------------------------------
// Core GEMM: C = A(bf16, MxK row-major) * B(bf16), fp32 accumulate via WMMA.
// BNK=1: B[k][n] = Bg[n*ldb + k]  (y = x @ W.T style)  -- async-staged
// BNK=0: B[k][n] = Bg[k*ldb + n]  (plain K x N row-major, e.g. V) -- LDS transpose
// A tile is double-buffered via GLOBAL_LOAD_ASYNC_TO_LDS_B128 (ASYNCcnt).
// Requires M%128==0, N%64==0, K%32==0 (true for every call here).
// ---------------------------------------------------------------------------
template <int BNK, int EPI>
__global__ __launch_bounds__(256) void gemm_wmma_kernel(
    const bf16_t* __restrict__ A, const bf16_t* __restrict__ Bm,
    void* __restrict__ Out,
    int M, int N, int K, int lda, int ldb,
    long strideA, long strideB, long strideOut,
    const float* __restrict__ bias,
    const float* __restrict__ keymask,
    float scale)
{
    __shared__ bf16_t As[2][BM * LSTR];
    __shared__ bf16_t Bs[2][BN * LSTR];

    const int z = blockIdx.z;
    const bf16_t* Ap = A + (long)z * strideA;
    const bf16_t* Bp = Bm + (long)z * strideB;

    const int m0 = blockIdx.y * BM;
    const int n0 = blockIdx.x * BN;

    const int tid  = threadIdx.x;
    const int lane = tid & 31;
    const int wv   = tid >> 5;      // 0..7
    const int wm   = wv >> 1;       // 0..3 -> 32-row band
    const int wn   = wv & 1;        // 0..1 -> 32-col band
    const int hsel = lane >> 4;     // lane half (ISA K-half select)
    const int l16  = lane & 15;

    union Frag { uint4 u[2]; bf16x16 v; };

    f32x8 acc[2][2] = {};

    // per-thread staging coordinates (one 16B async op each)
    const int ar  = tid >> 1;               // A row 0..127
    const int akh = (tid & 1) << 4;         // A k-offset 0/16
    const int bn_ = tid >> 1;               // B row (NK), valid when tid<128
    // async issue: HBM -> LDS directly, tracked by ASYNCcnt
    auto stage_async = [&](int k0, int buf) {
        {
            unsigned lds = (unsigned)(uintptr_t)&As[buf][ar * LSTR + akh];
            unsigned long long g =
                (unsigned long long)(uintptr_t)(Ap + (long)(m0 + ar) * lda + k0 + akh);
            asm volatile("global_load_async_to_lds_b128 %0, %1, off"
                         :: "v"(lds), "v"(g) : "memory");
        }
        if (BNK) {
            if (tid < 128) {                // waves 0..3 (wave-uniform)
                unsigned lds = (unsigned)(uintptr_t)&Bs[buf][bn_ * LSTR + akh];
                unsigned long long g =
                    (unsigned long long)(uintptr_t)(Bp + (long)(n0 + bn_) * ldb + k0 + akh);
                asm volatile("global_load_async_to_lds_b128 %0, %1, off"
                             :: "v"(lds), "v"(g) : "memory");
            }
        }
    };

    const int nIter = K / BK;
    stage_async(0, 0);

    for (int it = 0; it < nIter; ++it) {
        const int buf = it & 1;
        if (it + 1 < nIter) {
            stage_async((it + 1) * BK, buf ^ 1);
            // drain current stage only; next stage's (1 or 2) ops stay in flight
            if (BNK && wv < 4)
                asm volatile("s_wait_asynccnt 0x2" ::: "memory");
            else
                asm volatile("s_wait_asynccnt 0x1" ::: "memory");
        } else {
            asm volatile("s_wait_asynccnt 0x0" ::: "memory");
        }
        if (!BNK) {
            // stage B with transpose: Bs[n][k] = Bg[k*ldb + n]
            int k  = tid >> 3;               // 0..31
            int n8 = (tid & 7) << 3;         // 0..56
            uint4 v = *(const uint4*)&Bp[(long)(it * BK + k) * ldb + n0 + n8];
            const bf16_t* e = (const bf16_t*)&v;
            #pragma unroll
            for (int i = 0; i < 8; ++i)
                Bs[buf][(n8 + i) * LSTR + k] = e[i];
        }
        __syncthreads();   // stage `it` visible to all waves

        // ---- build fragments per ISA 16-bit A/B layouts ----
        Frag af[2], bfr[2];
        #pragma unroll
        for (int i = 0; i < 2; ++i) {
            int am = wm * 32 + i * 16 + l16;   // A row within block tile
            // lanes 0-15: K 0..7 & 16..23 ; lanes 16-31: K 8..15 & 24..31
            af[i].u[0] = *(const uint4*)&As[buf][am * LSTR + hsel * 8];
            af[i].u[1] = *(const uint4*)&As[buf][am * LSTR + 16 + hsel * 8];
            int bn = wn * 32 + i * 16 + l16;   // B col within block tile
            // lanes 0-15: K 0..7 & 16..23 ; lanes 16-31: K 8..15 & 24..31
            bfr[i].u[0] = *(const uint4*)&Bs[buf][bn * LSTR + hsel * 8];
            bfr[i].u[1] = *(const uint4*)&Bs[buf][bn * LSTR + 16 + hsel * 8];
        }
        #pragma unroll
        for (int i = 0; i < 2; ++i)
            #pragma unroll
            for (int j = 0; j < 2; ++j)
                acc[i][j] = __builtin_amdgcn_wmma_f32_16x16x32_bf16(
                    false, af[i].v, false, bfr[j].v, (short)0, acc[i][j],
                    false, false);
        __syncthreads();   // all reads of buf done before it is overwritten
    }

    // ---- epilogue: C/D layout is n = lane&15, m = r + 8*(lane>>4) ----
    #pragma unroll
    for (int i = 0; i < 2; ++i) {
        #pragma unroll
        for (int j = 0; j < 2; ++j) {
            #pragma unroll
            for (int r = 0; r < 8; ++r) {
                int row = m0 + wm * 32 + i * 16 + hsel * 8 + r;
                int col = n0 + wn * 32 + j * 16 + l16;
                float v = acc[i][j][r];
                if constexpr (EPI == EPI_SCORES) {
                    v *= scale;                                   // /sqrt(DH)
                    if (keymask[z * N + col] == 0.f) v = NEG_BIG; // key mask
                    if (col <= row) v += -10000.0f;               // temporal
                    ((float*)Out)[(long)z * strideOut + (long)row * N + col] = v;
                } else if constexpr (EPI == EPI_BF16_HEADS) {
                    int b = row >> 10, n = row & 1023;
                    int h = col >> 6,  dh = col & 63;
                    long dst = (((long)(h * B_SZ + b) * N_SZ) + n) * DH_SZ + dh;
                    ((bf16_t*)Out)[dst] = (bf16_t)v;
                } else if constexpr (EPI == EPI_F32_HEADMERGE) {
                    int h = z >> 2, b = z & 3;                    // z = h*B + b
                    long dst = ((long)(b * N_SZ + row)) * D_SZ + h * DH_SZ + col;
                    ((float*)Out)[dst] = v;
                } else if constexpr (EPI == EPI_BF16_BIAS_RELU) {
                    v += bias[col];
                    v = v > 0.f ? v : 0.f;
                    ((bf16_t*)Out)[(long)row * N + col] = (bf16_t)v;
                } else if constexpr (EPI == EPI_F32_BIAS) {
                    v += bias[col];
                    ((float*)Out)[(long)row * N + col] = v;
                } else {
                    ((bf16_t*)Out)[(long)row * N + col] = (bf16_t)v;
                }
            }
        }
    }
}

// ---------------------------------------------------------------------------
// Elementwise / reduction helpers
// ---------------------------------------------------------------------------
__global__ void cvt_bf16_kernel(const float* __restrict__ in,
                                bf16_t* __restrict__ out, long n) {
    long i = (long)blockIdx.x * blockDim.x + threadIdx.x;
    if (i < n) out[i] = (bf16_t)in[i];
}

// sign(sum |X[z,n,:]|) over DH -> {0,1}
__global__ void absmask_kernel(const bf16_t* __restrict__ X,
                               float* __restrict__ msk, int rows) {
    int id = blockIdx.x * blockDim.x + threadIdx.x;
    if (id >= rows) return;
    const bf16_t* p = X + (long)id * DH_SZ;
    float s = 0.f;
    #pragma unroll 8
    for (int i = 0; i < DH_SZ; ++i) s += fabsf((float)p[i]);
    msk[id] = (s > 0.f) ? 1.f : 0.f;
}

// softmax over axis 0 (z = head*B + b), then * query_mask[z,n]; emit bf16 A
__global__ void softmax_z_kernel(const float* __restrict__ S,
                                 const float* __restrict__ qmask,
                                 bf16_t* __restrict__ attn) {
    const long NM = (long)N_SZ * N_SZ;
    long id = (long)blockIdx.x * blockDim.x + threadIdx.x;
    if (id >= NM) return;
    int n = (int)(id >> 10);                    // query row
    float mx = -3.402823466e38f;
    for (int z = 0; z < Z_SZ; ++z) mx = fmaxf(mx, S[(long)z * NM + id]);
    float sum = 0.f;
    for (int z = 0; z < Z_SZ; ++z) sum += __expf(S[(long)z * NM + id] - mx);
    float inv = 1.f / sum;
    for (int z = 0; z < Z_SZ; ++z) {
        float a = __expf(S[(long)z * NM + id] - mx) * inv * qmask[z * N_SZ + n];
        attn[(long)z * NM + id] = (bf16_t)a;
    }
}

__device__ __forceinline__ float block_sum(float v, float* red) {
    int tid = threadIdx.x;
    red[tid] = v; __syncthreads();
    for (int s = 128; s > 0; s >>= 1) {
        if (tid < s) red[tid] += red[tid + s];
        __syncthreads();
    }
    float r = red[0]; __syncthreads();
    return r;
}

// h = LN(x + attn_out*valid) ; also emit bf16 copy for the next GEMM
__global__ __launch_bounds__(256) void add_ln_kernel(
    const float* __restrict__ x, const float* __restrict__ ao,
    const unsigned char* __restrict__ kpm,
    const float* __restrict__ g, const float* __restrict__ beta,
    float* __restrict__ hout, bf16_t* __restrict__ hb)
{
    __shared__ float red[256];
    int row = blockIdx.x;                       // 0..4095 = b*N + n
    float valid = kpm[row] ? 0.f : 1.f;
    long base = (long)row * D_SZ;
    float vals[4]; float s = 0.f;
    #pragma unroll
    for (int i = 0; i < 4; ++i) {
        int c = i * 256 + threadIdx.x;
        vals[i] = x[base + c] + ao[base + c] * valid;
        s += vals[i];
    }
    float mean = block_sum(s, red) * (1.0f / D_SZ);
    float vs = 0.f;
    #pragma unroll
    for (int i = 0; i < 4; ++i) { float d = vals[i] - mean; vs += d * d; }
    float var = block_sum(vs, red) * (1.0f / D_SZ);
    float rstd = rsqrtf(var + 1e-5f);
    #pragma unroll
    for (int i = 0; i < 4; ++i) {
        int c = i * 256 + threadIdx.x;
        float o = (vals[i] - mean) * rstd * g[c] + beta[c];
        hout[base + c] = o;
        hb[base + c] = (bf16_t)o;
    }
}

// out = LN(ff*valid + h) * valid
__global__ __launch_bounds__(256) void final_ln_kernel(
    const float* __restrict__ ff, const float* __restrict__ h,
    const unsigned char* __restrict__ kpm,
    const float* __restrict__ g, const float* __restrict__ beta,
    float* __restrict__ out)
{
    __shared__ float red[256];
    int row = blockIdx.x;
    float valid = kpm[row] ? 0.f : 1.f;
    long base = (long)row * D_SZ;
    float vals[4]; float s = 0.f;
    #pragma unroll
    for (int i = 0; i < 4; ++i) {
        int c = i * 256 + threadIdx.x;
        vals[i] = ff[base + c] * valid + h[base + c];
        s += vals[i];
    }
    float mean = block_sum(s, red) * (1.0f / D_SZ);
    float vs = 0.f;
    #pragma unroll
    for (int i = 0; i < 4; ++i) { float d = vals[i] - mean; vs += d * d; }
    float var = block_sum(vs, red) * (1.0f / D_SZ);
    float rstd = rsqrtf(var + 1e-5f);
    #pragma unroll
    for (int i = 0; i < 4; ++i) {
        int c = i * 256 + threadIdx.x;
        float o = (vals[i] - mean) * rstd * g[c] + beta[c];
        out[base + c] = o * valid;
    }
}

// ---------------------------------------------------------------------------
extern "C" void kernel_launch(void* const* d_in, const int* in_sizes, int n_in,
                              void* d_out, int out_size, void* d_ws, size_t ws_size,
                              hipStream_t stream)
{
    (void)in_sizes; (void)n_in; (void)out_size; (void)ws_size;
    const float* x   = (const float*)d_in[0];
    const unsigned char* kpm = (const unsigned char*)d_in[1];   // bool mask
    const float* Wq  = (const float*)d_in[2];
    const float* Wk  = (const float*)d_in[3];
    const float* Wv  = (const float*)d_in[4];
    const float* W1  = (const float*)d_in[5];
    const float* b1  = (const float*)d_in[6];
    const float* W2  = (const float*)d_in[7];
    const float* b2  = (const float*)d_in[8];
    const float* g1  = (const float*)d_in[9];
    const float* be1 = (const float*)d_in[10];
    const float* g2  = (const float*)d_in[11];
    const float* be2 = (const float*)d_in[12];

    char* wsp = (char*)d_ws;
    size_t off = 0;
    auto alloc = [&](size_t bytes) -> void* {
        void* p = wsp + off;
        off = (off + bytes + 255) & ~(size_t)255;
        return p;
    };
    const long MD  = (long)MROWS * D_SZ;          // 4096*1024
    const long DD  = (long)D_SZ * D_SZ;
    const long ZND = (long)Z_SZ * N_SZ * DH_SZ;   // 64*1024*64
    const long NM  = (long)N_SZ * N_SZ;           // 1M

    bf16_t* xb    = (bf16_t*)alloc(MD * 2);
    bf16_t* Wqb   = (bf16_t*)alloc(DD * 2);
    bf16_t* Wkb   = (bf16_t*)alloc(DD * 2);
    bf16_t* Wvb   = (bf16_t*)alloc(DD * 2);
    bf16_t* W1b   = (bf16_t*)alloc(DD * 2);
    bf16_t* W2b   = (bf16_t*)alloc(DD * 2);
    bf16_t* Qb    = (bf16_t*)alloc(ZND * 2);
    bf16_t* Kb    = (bf16_t*)alloc(ZND * 2);
    bf16_t* Vb    = (bf16_t*)alloc(ZND * 2);
    float*  qmask = (float*)alloc((long)Z_SZ * N_SZ * 4);
    float*  kmask = (float*)alloc((long)Z_SZ * N_SZ * 4);
    float*  S     = (float*)alloc((long)Z_SZ * NM * 4);   // 256 MB
    bf16_t* attnb = (bf16_t*)alloc((long)Z_SZ * NM * 2);  // 128 MB
    float*  ao    = (float*)alloc(MD * 4);
    float*  hbuf  = (float*)alloc(MD * 4);
    bf16_t* hb    = (bf16_t*)alloc(MD * 2);
    bf16_t* ff1b  = (bf16_t*)alloc(MD * 2);
    float*  ffbuf = (float*)alloc(MD * 4);

    // fp32 -> bf16 operand conversion
    cvt_bf16_kernel<<<(int)((MD + 255) / 256), 256, 0, stream>>>(x,  xb,  MD);
    cvt_bf16_kernel<<<(int)((DD + 255) / 256), 256, 0, stream>>>(Wq, Wqb, DD);
    cvt_bf16_kernel<<<(int)((DD + 255) / 256), 256, 0, stream>>>(Wk, Wkb, DD);
    cvt_bf16_kernel<<<(int)((DD + 255) / 256), 256, 0, stream>>>(Wv, Wvb, DD);
    cvt_bf16_kernel<<<(int)((DD + 255) / 256), 256, 0, stream>>>(W1, W1b, DD);
    cvt_bf16_kernel<<<(int)((DD + 255) / 256), 256, 0, stream>>>(W2, W2b, DD);

    // QKV projections: (4096x1024)@(1024x1024).T -> head-major bf16 [64,1024,64]
    dim3 gproj(N_SZ / BN, MROWS / BM, 1);
    gemm_wmma_kernel<1, EPI_BF16_HEADS><<<gproj, 256, 0, stream>>>(
        xb, Wqb, Qb, MROWS, D_SZ, D_SZ, D_SZ, D_SZ, 0, 0, 0,
        nullptr, nullptr, 1.0f);
    gemm_wmma_kernel<1, EPI_BF16_HEADS><<<gproj, 256, 0, stream>>>(
        xb, Wkb, Kb, MROWS, D_SZ, D_SZ, D_SZ, D_SZ, 0, 0, 0,
        nullptr, nullptr, 1.0f);
    gemm_wmma_kernel<1, EPI_BF16_HEADS><<<gproj, 256, 0, stream>>>(
        xb, Wvb, Vb, MROWS, D_SZ, D_SZ, D_SZ, D_SZ, 0, 0, 0,
        nullptr, nullptr, 1.0f);

    // sign(sum|Q|), sign(sum|K|)
    absmask_kernel<<<(Z_SZ * N_SZ + 255) / 256, 256, 0, stream>>>(Qb, qmask, Z_SZ * N_SZ);
    absmask_kernel<<<(Z_SZ * N_SZ + 255) / 256, 256, 0, stream>>>(Kb, kmask, Z_SZ * N_SZ);

    // scores[z] = Q[z] @ K[z].T / 8, + key mask + temporal mask (fused epilogue)
    dim3 gsc(N_SZ / BN, N_SZ / BM, Z_SZ);
    gemm_wmma_kernel<1, EPI_SCORES><<<gsc, 256, 0, stream>>>(
        Qb, Kb, S, N_SZ, N_SZ, DH_SZ, DH_SZ, DH_SZ,
        (long)N_SZ * DH_SZ, (long)N_SZ * DH_SZ, NM,
        nullptr, kmask, 0.125f);

    // softmax over z-axis (faithful to torch implicit dim=0), * query_mask -> bf16
    softmax_z_kernel<<<(int)((NM + 255) / 256), 256, 0, stream>>>(S, qmask, attnb);

    // out[z] = attn[z] @ V[z] -> merged back to (B,N,D) fp32
    dim3 gav(DH_SZ / BN, N_SZ / BM, Z_SZ);
    gemm_wmma_kernel<0, EPI_F32_HEADMERGE><<<gav, 256, 0, stream>>>(
        attnb, Vb, ao, N_SZ, DH_SZ, N_SZ, N_SZ, DH_SZ,
        NM, (long)N_SZ * DH_SZ, 0,
        nullptr, nullptr, 1.0f);

    // h = LN(x + attn_out * valid)
    add_ln_kernel<<<MROWS, 256, 0, stream>>>(x, ao, kpm, g1, be1, hbuf, hb);

    // FFN: relu(h@W1.T + b1) @ W2.T + b2
    dim3 gffn(D_SZ / BN, MROWS / BM, 1);
    gemm_wmma_kernel<1, EPI_BF16_BIAS_RELU><<<gffn, 256, 0, stream>>>(
        hb, W1b, ff1b, MROWS, D_SZ, D_SZ, D_SZ, D_SZ, 0, 0, 0,
        b1, nullptr, 1.0f);
    gemm_wmma_kernel<1, EPI_F32_BIAS><<<gffn, 256, 0, stream>>>(
        ff1b, W2b, ffbuf, MROWS, D_SZ, D_SZ, D_SZ, D_SZ, 0, 0, 0,
        b2, nullptr, 1.0f);

    // out = LN(ff*valid + h) * valid
    final_ln_kernel<<<MROWS, 256, 0, stream>>>(ffbuf, hbuf, kpm, g2, be2,
                                               (float*)d_out);
}
